// SimpleRNN_35081292874338
// MI455X (gfx1250) — matrix operand order repaired and checked by reading, not compiled, not run on hardware
//
#include <hip/hip_runtime.h>
#include <hip/hip_bf16.h>

typedef __attribute__((ext_vector_type(16))) _Float16 v16h;
typedef __attribute__((ext_vector_type(8)))  float    v8f;

#define VOCAB   32000
#define EMBED   256
#define HIDDEN  256
#define NBATCH  128
#define SEQ     2048
#define KCH     8          // 256 / 32 K-chunks per WMMA sweep
#define ROWPAD  264        // halves per LDS row: 256 + 8 pad (132 dwords -> bank spread)

#if __has_builtin(__builtin_amdgcn_tanhf)
#define FAST_TANH(x) __builtin_amdgcn_tanhf(x)
#else
#define FAST_TANH(x) tanhf(x)
#endif

union AFrag { v16h v; uint4 q[2]; };
union XPack { uint4 q; _Float16 h[8]; };

__global__ __launch_bounds__(512)
void rnn_fused_wmma(const int* __restrict__ X,      // (N, L) int32
                    const float* __restrict__ emb,   // (VOCAB, E)
                    const float* __restrict__ W_hh,  // (H, H)
                    const float* __restrict__ b_hh,  // (H)
                    const float* __restrict__ W_xh,  // (E, H)
                    const float* __restrict__ b_xh,  // (H)
                    float* __restrict__ out)         // (N, H)
{
    __shared__ __align__(16) _Float16 hbuf[2][16 * ROWPAD];  // ping-pong hidden state
    __shared__ __align__(16) _Float16 xebuf[2][16 * ROWPAD]; // ping-pong embeddings

    const int wg   = blockIdx.x;           // 8 blocks: 16 batch rows each
    const int wave = threadIdx.x >> 5;     // 16 waves: one 16-col N-tile each
    const int lane = threadIdx.x & 31;
    const int mrow = lane & 15;            // A-matrix row / C col-index
    const int khi  = lane >> 4;            // lane half selects K window
    const int ncol = (wave << 4) + mrow;   // output column owned by this lane

    // ---- Preload B fragments (columns ncol of W_xh and W_hh, transposed, f16).
    // B 32x16 layout: lane = N col, element j holds K = khi*8 + j (j<8), 16 + khi*8 + (j-8) (j>=8) within chunk.
    v16h Bx[KCH], Bh[KCH];
    {
        const int kb0 = khi * 8;
        #pragma unroll
        for (int c = 0; c < KCH; ++c) {
            const int kb = c * 32 + kb0;
            v16h bx, bh;
            #pragma unroll
            for (int j = 0; j < 16; ++j) {
                bx[j] = (_Float16)W_xh[(size_t)(kb + j + (j & 8)) * HIDDEN + ncol];
                bh[j] = (_Float16)W_hh[(size_t)(kb + j + (j & 8)) * HIDDEN + ncol];
            }
            Bx[c] = bx; Bh[c] = bh;
        }
    }
    const float bias = b_xh[ncol] + b_hh[ncol];

    // h(0) = 0
    for (int i = threadIdx.x; i < 16 * ROWPAD; i += 512)
        hbuf[0][i] = (_Float16)0.f;

    const int nrow = wg * 16 + wave;       // wave w gathers batch row w

    // Prologue: gather + commit xe(0); start gather of xe(1).
    float4 g0, g1;
    {
        const int id = X[(size_t)nrow * SEQ + 0];
        const float4* ep = (const float4*)(emb + (size_t)id * EMBED);
        g0 = ep[lane * 2 + 0];
        g1 = ep[lane * 2 + 1];
        XPack u;
        u.h[0] = (_Float16)g0.x; u.h[1] = (_Float16)g0.y;
        u.h[2] = (_Float16)g0.z; u.h[3] = (_Float16)g0.w;
        u.h[4] = (_Float16)g1.x; u.h[5] = (_Float16)g1.y;
        u.h[6] = (_Float16)g1.z; u.h[7] = (_Float16)g1.w;
        *(uint4*)&xebuf[0][wave * ROWPAD + lane * 8] = u.q;
    }
    {
        const int id = X[(size_t)nrow * SEQ + 1];
        const float4* ep = (const float4*)(emb + (size_t)id * EMBED);
        g0 = ep[lane * 2 + 0];
        g1 = ep[lane * 2 + 1];
    }
    __syncthreads();

    float tv[8];
    const int aoff = khi * 8;
    const int arow = mrow * ROWPAD;

    for (int t = 0; t < SEQ; ++t) {
        const _Float16* xc = xebuf[t & 1];
        const _Float16* hc = hbuf[t & 1];

        v8f accx = { bias, bias, bias, bias, bias, bias, bias, bias };
        v8f acch = {};

        // Software-pipelined A-fragment loads: chunk c+1 issued before chunk c's
        // WMMAs so the scheduler can use partial dscnt waits.
        AFrag ax[2], ah[2];
        ax[0].q[0] = *(const uint4*)&xc[arow + aoff];
        ax[0].q[1] = *(const uint4*)&xc[arow + aoff + 16];
        ah[0].q[0] = *(const uint4*)&hc[arow + aoff];
        ah[0].q[1] = *(const uint4*)&hc[arow + aoff + 16];
        #pragma unroll
        for (int c = 0; c < KCH; ++c) {
            const int cb = c & 1, nb = (c + 1) & 1;
            if (c + 1 < KCH) {
                const int base = (c + 1) * 32 + aoff;
                ax[nb].q[0] = *(const uint4*)&xc[arow + base];
                ax[nb].q[1] = *(const uint4*)&xc[arow + base + 16];
                ah[nb].q[0] = *(const uint4*)&hc[arow + base];
                ah[nb].q[1] = *(const uint4*)&hc[arow + base + 16];
            }
            accx = __builtin_amdgcn_wmma_f32_16x16x32_f16(
                       false, ax[cb].v, false, Bx[c], (short)0, accx, false, false);
            acch = __builtin_amdgcn_wmma_f32_16x16x32_f16(
                       false, ah[cb].v, false, Bh[c], (short)0, acch, false, false);
        }

        // h(t+1) = tanh(acc); C/D layout: VGPR r -> row r + 8*khi, lane&15 -> col.
        _Float16* hn = hbuf[(t + 1) & 1];
        #pragma unroll
        for (int r = 0; r < 8; ++r) {
            const float v = FAST_TANH(accx[r] + acch[r]);
            tv[r] = v;
            hn[(r + 8 * khi) * ROWPAD + ncol] = (_Float16)v;
        }

        // Commit xe(t+1) (gathered one full iteration ago) into the other buffer.
        if (t + 1 < SEQ) {
            XPack u;
            u.h[0] = (_Float16)g0.x; u.h[1] = (_Float16)g0.y;
            u.h[2] = (_Float16)g0.z; u.h[3] = (_Float16)g0.w;
            u.h[4] = (_Float16)g1.x; u.h[5] = (_Float16)g1.y;
            u.h[6] = (_Float16)g1.z; u.h[7] = (_Float16)g1.w;
            *(uint4*)&xebuf[(t + 1) & 1][wave * ROWPAD + lane * 8] = u.q;
        }
        // Start gather of xe(t+2); consumed next iteration.
        if (t + 2 < SEQ) {
            const int id = X[(size_t)nrow * SEQ + (t + 2)];
            const float4* ep = (const float4*)(emb + (size_t)id * EMBED);
            g0 = ep[lane * 2 + 0];
            g1 = ep[lane * 2 + 1];
        }

        __syncthreads();   // single barrier per step: all writes went to opposite buffers
    }

    // Final hidden state, f32.
    #pragma unroll
    for (int r = 0; r < 8; ++r)
        out[(size_t)(wg * 16 + r + 8 * khi) * HIDDEN + ncol] = tv[r];
}

extern "C" void kernel_launch(void* const* d_in, const int* in_sizes, int n_in,
                              void* d_out, int out_size, void* d_ws, size_t ws_size,
                              hipStream_t stream) {
    (void)in_sizes; (void)n_in; (void)d_ws; (void)ws_size; (void)out_size;
    const int*   X    = (const int*)  d_in[0];
    const float* emb  = (const float*)d_in[1];
    const float* W_hh = (const float*)d_in[2];
    const float* b_hh = (const float*)d_in[3];
    const float* W_xh = (const float*)d_in[4];
    const float* b_xh = (const float*)d_in[5];
    float* out = (float*)d_out;

    dim3 grid(NBATCH / 16);   // 8 workgroups, each owns 16 independent batch rows
    dim3 block(512);          // 16 wave32s, each owns one 16-column N-tile
    rnn_fused_wmma<<<grid, block, 0, stream>>>(X, emb, W_hh, b_hh, W_xh, b_xh, out);
}
